// FSModel_549755814271
// MI455X (gfx1250) — matrix operand order repaired and checked by reading, not compiled, not run on hardware
//
#include <hip/hip_runtime.h>

// ---------------------------------------------------------------------------
// FastSpeech-like forward for MI455X (gfx1250, wave32, WMMA).
// Live dataflow only (MHA + duration predictor outputs are discarded in the
// reference). Convs (K=3, SAME) = 3 shifted GEMMs accumulated in registers
// via v_wmma_f32_16x16x32_bf16 (bf16 A/B, f32 accumulate).
//
// bf16 activation buffers use a guard-row layout [B][Len+2][C] (zeroed edge
// rows) so conv taps never need per-lane predication: the GEMM k-loop is a
// single branchless block of constant-offset b128 loads + WMMAs.
// ---------------------------------------------------------------------------

typedef unsigned int u32x4 __attribute__((ext_vector_type(4)));
typedef __bf16 bf16x16 __attribute__((ext_vector_type(16)));
typedef float f32x8 __attribute__((ext_vector_type(8)));

union Frag { bf16x16 f; u32x4 q[2]; };

#define BB 32
#define TT 256
#define DD 384
#define LL 4
#define C1 1536
#define MAXMEL 1280
#define NOUT 80

// ---- weight conversion: Wc1 [L,C1,D,3] f32 -> [L,3,C1,D] bf16 -------------
__global__ void cvt_w1(const float* __restrict__ w, __bf16* __restrict__ o, int total) {
    int idx = blockIdx.x * blockDim.x + threadIdx.x;
    if (idx >= total) return;
    int ci = idx % DD; int r = idx / DD;
    int co = r % C1;   r /= C1;
    int t  = r % 3;    int i = r / 3;
    o[idx] = (__bf16)w[(((size_t)i * C1 + co) * DD + ci) * 3 + t];
}
// Wc2 [L,D,C1,3] f32 -> [L,3,D,C1] bf16
__global__ void cvt_w2(const float* __restrict__ w, __bf16* __restrict__ o, int total) {
    int idx = blockIdx.x * blockDim.x + threadIdx.x;
    if (idx >= total) return;
    int ci = idx % C1; int r = idx / C1;
    int co = r % DD;   r /= DD;
    int t  = r % 3;    int i = r / 3;
    o[idx] = (__bf16)w[(((size_t)i * DD + co) * C1 + ci) * 3 + t];
}
// Wout [D,80] f32 -> transposed, zero-padded [128][D] bf16
__global__ void cvt_wout(const float* __restrict__ w, __bf16* __restrict__ o) {
    int idx = blockIdx.x * blockDim.x + threadIdx.x;
    if (idx >= 128 * DD) return;
    int k = idx % DD; int n = idx / DD;
    o[idx] = (n < NOUT) ? (__bf16)w[(size_t)k * NOUT + n] : (__bf16)0.0f;
}

// ---- zero the guard rows of a padded bf16 activation buffer ---------------
__global__ void zero_guards(__bf16* __restrict__ buf, int Len, int C, int total) {
    int idx = blockIdx.x * blockDim.x + threadIdx.x;   // total = BB*2*C
    if (idx >= total) return;
    int c = idx % C; int r = idx / C;
    int which = r & 1; int b = r >> 1;
    size_t row = (size_t)b * (Len + 2) + (which ? (Len + 1) : 0);
    buf[row * C + c] = (__bf16)0.0f;
}

// ---- embedding + positional encoding (PE indexed by BATCH, per source bug)
__global__ void embed_pe(const int* __restrict__ tok, const float* __restrict__ emb,
                         float* __restrict__ xf, __bf16* __restrict__ xb) {
    int idx = blockIdx.x * blockDim.x + threadIdx.x;
    if (idx >= BB * TT * DD) return;
    int d = idx % DD; int r = idx / DD;
    int t = r % TT;   int b = r / TT;
    int tk = tok[b * TT + t];
    float v = emb[(size_t)tk * DD + d];
    int j = d >> 1;
    float divj = __expf(-(float)(2 * j) * (9.210340371976184f / (float)DD)); // ln(1e4)/D
    float ang = (float)b * divj;
    v += (d & 1) ? __cosf(ang) : __sinf(ang);
    xf[idx] = v;
    xb[((size_t)(b * (TT + 2) + 1 + t)) * DD + d] = (__bf16)v;   // padded layout
}

// ---- WMMA conv-as-GEMM -----------------------------------------------------
// A: bf16 activations, guard-padded [B][Len+2][CIN]; W: bf16 [NTAPS][COUT][CIN].
// Wave computes 32x64 of C (8 accumulators); block tile 128x64.
// MODE 0: bias+ReLU -> bf16 padded rows (stride COUT)
// MODE 1: bias -> f32 compact rows (stride COUT)
// MODE 2: bias -> f32 compact rows, store only n<CoutReal, stride CoutReal
template <int CIN, int COUT, int NTAPS, int MODE>
__global__ __launch_bounds__(128) void gemm_wmma(
    const __bf16* __restrict__ A, const __bf16* __restrict__ W,
    const float* __restrict__ bias, void* __restrict__ Out,
    int Len, int CoutReal)
{
    const int lane = threadIdx.x & 31;
    const int wv   = threadIdx.x >> 5;
    const int hi   = lane >> 4;     // half-wave selects K sub-range (ISA layout)
    const int lm   = lane & 15;
    const int m0   = blockIdx.x * 128 + wv * 32;   // wave's first compact row
    const int row  = m0 + lm;
    const int b    = row / Len;                    // 32-row tile stays in one batch
    const int p    = row + 2 * b + 1;              // padded row index
    const int nBase = blockIdx.y * 64;

    // A: VGPR0-3 = K[hi*8..+8), VGPR4-7 = K[16+hi*8..+8)   (16-bit A 16x32)
    // B: VGPR0-3 = K[hi*8..+8), VGPR4-7 = K[16+hi*8..+8) of column n (matches A)
    const __bf16* aptr = A + (size_t)p * CIN + hi * 8;
    const __bf16* bptr = W + (size_t)(nBase + lm) * CIN + hi * 8;

    f32x8 acc[8];
#pragma unroll
    for (int s = 0; s < 8; ++s) { f32x8 z = {}; acc[s] = z; }

    for (int k0 = 0; k0 < CIN; k0 += 32) {
#pragma unroll
        for (int t = 0; t < NTAPS; ++t) {
            const int aoff = (t - NTAPS / 2) * CIN;        // constant row shift
            Frag a0, a1;
            a0.q[0] = *(const u32x4*)(aptr + aoff);
            a0.q[1] = *(const u32x4*)(aptr + aoff + 16);
            a1.q[0] = *(const u32x4*)(aptr + aoff + 16 * CIN);
            a1.q[1] = *(const u32x4*)(aptr + aoff + 16 * CIN + 16);
            const __bf16* bt = bptr + (size_t)t * COUT * CIN;  // constant tap stride
#pragma unroll
            for (int s = 0; s < 4; ++s) {
                Frag bf_;
                bf_.q[0] = *(const u32x4*)(bt + s * 16 * CIN);
                bf_.q[1] = *(const u32x4*)(bt + s * 16 * CIN + 16);
                acc[s] = __builtin_amdgcn_wmma_f32_16x16x32_bf16(
                    false, a0.f, false, bf_.f, (short)0, acc[s], false, false);
                acc[4 + s] = __builtin_amdgcn_wmma_f32_16x16x32_bf16(
                    false, a1.f, false, bf_.f, (short)0, acc[4 + s], false, false);
            }
        }
        aptr += 32;
        bptr += 32;
    }

#pragma unroll
    for (int h = 0; h < 2; ++h) {
#pragma unroll
        for (int s = 0; s < 4; ++s) {
            const int n = nBase + s * 16 + lm;
            const float bv = (n < CoutReal) ? bias[n] : 0.0f;
#pragma unroll
            for (int j = 0; j < 8; ++j) {
                const int m = m0 + h * 16 + hi * 8 + j;    // compact row (same batch b)
                float v = acc[h * 4 + s][j] + bv;
                if (MODE == 0) {
                    v = v > 0.0f ? v : 0.0f;
                    ((__bf16*)Out)[(size_t)(m + 2 * b + 1) * COUT + n] = (__bf16)v;
                } else if (MODE == 1) {
                    ((float*)Out)[(size_t)m * COUT + n] = v;
                } else {
                    if (n < CoutReal) ((float*)Out)[(size_t)m * CoutReal + n] = v;
                }
            }
        }
    }
}

// ---- residual add + LayerNorm (one block per row, D=384, 128 thr x 3) -----
// f32 in/out compact; bf16 out in guard-padded layout for the next GEMM.
__global__ __launch_bounds__(128) void residual_ln(
    const float* __restrict__ X, const float* __restrict__ Y,
    const float* __restrict__ g, const float* __restrict__ bta,
    float* __restrict__ Xo, __bf16* __restrict__ Xb, int Len)
{
    const int row = blockIdx.x;
    const int tid = threadIdx.x;
    __shared__ float sS[128], sQ[128];
    const size_t base = (size_t)row * DD;
    const size_t pbase = (size_t)(row + 2 * (row / Len) + 1) * DD;
    float v[3]; float s = 0.0f, q = 0.0f;
#pragma unroll
    for (int k = 0; k < 3; ++k) {
        int d = tid + k * 128;
        float xv = X[base + d] + Y[base + d];
        v[k] = xv; s += xv; q += xv * xv;
    }
    sS[tid] = s; sQ[tid] = q; __syncthreads();
    for (int st = 64; st > 0; st >>= 1) {
        if (tid < st) { sS[tid] += sS[tid + st]; sQ[tid] += sQ[tid + st]; }
        __syncthreads();
    }
    float mean = sS[0] * (1.0f / DD);
    float var  = sQ[0] * (1.0f / DD) - mean * mean;
    float rstd = rsqrtf(var + 1e-5f);
#pragma unroll
    for (int k = 0; k < 3; ++k) {
        int d = tid + k * 128;
        float o = (v[k] - mean) * rstd * g[d] + bta[d];
        Xo[base + d] = o;
        Xb[pbase + d] = (__bf16)o;
    }
}

// ---- length regulator ------------------------------------------------------
__global__ void cumsum_k(const int* __restrict__ dur, int* __restrict__ cs) {
    int b = threadIdx.x;
    if (b >= BB) return;
    int s = 0;
    for (int t = 0; t < TT; ++t) { s += dur[b * TT + t]; cs[b * TT + t] = s; }
}

__global__ __launch_bounds__(128) void regulate(
    const float* __restrict__ X, const int* __restrict__ CS,
    float* __restrict__ Mf, __bf16* __restrict__ Mb)
{
    const int bm = blockIdx.x;
    const int b = bm / MAXMEL;
    const int m = bm - b * MAXMEL;
    const int* cs = CS + b * TT;
    const int total = cs[TT - 1];
    const bool inr = (m < total);
    int t = 0;
    if (inr) {  // smallest t with cs[t] > m  => token whose span covers m
        int lo = 0, hiB = TT - 1;
        while (lo < hiB) { int mid = (lo + hiB) >> 1; if (cs[mid] > m) hiB = mid; else lo = mid + 1; }
        t = lo;
    }
    const size_t src = ((size_t)b * TT + t) * DD;
    const size_t dst = (size_t)bm * DD;
    const size_t pdst = (size_t)(bm + 2 * b + 1) * DD;
#pragma unroll
    for (int k = 0; k < 3; ++k) {
        int d = threadIdx.x + k * 128;
        float val = inr ? X[src + d] : 0.0f;
        Mf[dst + d] = val;
        Mb[pdst + d] = (__bf16)val;
    }
}

__global__ void copy_dur(const int* __restrict__ dur, float* __restrict__ o, int n) {
    int idx = blockIdx.x * blockDim.x + threadIdx.x;
    if (idx < n) o[idx] = (float)dur[idx];
}

// ---------------------------------------------------------------------------
extern "C" void kernel_launch(void* const* d_in, const int* in_sizes, int n_in,
                              void* d_out, int out_size, void* d_ws, size_t ws_size,
                              hipStream_t stream)
{
    (void)n_in; (void)out_size; (void)ws_size;
    const int* tokens = (const int*)d_in[0];
    const int* durs   = (const int*)d_in[2];

    // Param flattening: insertion order (emb first -> in_sizes[3]==15360)
    // or jax-tree sorted keys (Wc1 first).
    const float *emb, *Wc1, *bc1, *Wc2, *bc2, *ng, *nb, *Wout, *bout;
    if (in_sizes[3] == 15360) {
        emb  = (const float*)d_in[3];
        Wc1  = (const float*)d_in[12]; bc1 = (const float*)d_in[13];
        Wc2  = (const float*)d_in[14]; bc2 = (const float*)d_in[15];
        ng   = (const float*)d_in[16]; nb  = (const float*)d_in[17];
        Wout = (const float*)d_in[28]; bout = (const float*)d_in[29];
    } else {
        Wc1  = (const float*)d_in[3];  Wc2 = (const float*)d_in[4];
        Wout = (const float*)d_in[6];
        bc1  = (const float*)d_in[10]; bc2 = (const float*)d_in[11];
        bout = (const float*)d_in[13];
        emb  = (const float*)d_in[26];
        nb   = (const float*)d_in[27]; ng  = (const float*)d_in[28];
    }
    float* outF = (float*)d_out;

    char* base = (char*)d_ws;
    size_t off = 0;
    auto carve = [&](size_t bytes) -> void* {
        off = (off + 255) & ~(size_t)255;
        void* r = base + off;
        off += bytes;
        return r;
    };

    const size_t EW1  = (size_t)LL * 3 * C1 * DD;   // 7,077,888
    const size_t EW2  = (size_t)LL * 3 * DD * C1;
    const size_t EWO  = (size_t)128 * DD;
    const size_t ENC  = (size_t)BB * TT * DD;
    const size_t MEL  = (size_t)BB * MAXMEL * DD;
    const size_t ENCP = (size_t)BB * (TT + 2) * DD;        // guard-padded bf16
    const size_t MELP = (size_t)BB * (MAXMEL + 2) * DD;
    const size_t Y1P  = (size_t)BB * (MAXMEL + 2) * C1;

    __bf16* w1t  = (__bf16*)carve(EW1 * 2);
    __bf16* w2t  = (__bf16*)carve(EW2 * 2);
    __bf16* wot  = (__bf16*)carve(EWO * 2);
    float*  encF = (float*) carve(ENC * 4);
    __bf16* encB = (__bf16*)carve(ENCP * 2);
    float*  melF = (float*) carve(MEL * 4);
    __bf16* melB = (__bf16*)carve(MELP * 2);
    __bf16* y1B  = (__bf16*)carve(Y1P * 2);
    float*  y2F  = (float*) carve(MEL * 4);
    int*    cs   = (int*)   carve((size_t)BB * TT * 4);

    cvt_w1<<<(unsigned)((EW1 + 255) / 256), 256, 0, stream>>>(Wc1, w1t, (int)EW1);
    cvt_w2<<<(unsigned)((EW2 + 255) / 256), 256, 0, stream>>>(Wc2, w2t, (int)EW2);
    cvt_wout<<<(unsigned)((EWO + 255) / 256), 256, 0, stream>>>(Wout, wot);
    zero_guards<<<(BB * 2 * DD + 255) / 256, 256, 0, stream>>>(encB, TT, DD, BB * 2 * DD);
    zero_guards<<<(BB * 2 * C1 + 255) / 256, 256, 0, stream>>>(y1B, TT, C1, BB * 2 * C1);
    embed_pe<<<(unsigned)((ENC + 255) / 256), 256, 0, stream>>>(tokens, emb, encF, encB);

    // ---- encoder: 4x (conv1 -> relu -> conv2 -> residual LN) ----
    for (int i = 0; i < LL; ++i) {
        dim3 g1(BB * TT / 128, C1 / 64);
        gemm_wmma<DD, C1, 3, 0><<<g1, 128, 0, stream>>>(
            encB, w1t + (size_t)i * 3 * C1 * DD, bc1 + i * C1, (void*)y1B, TT, C1);
        dim3 g2(BB * TT / 128, DD / 64);
        gemm_wmma<C1, DD, 3, 1><<<g2, 128, 0, stream>>>(
            y1B, w2t + (size_t)i * 3 * DD * C1, bc2 + i * DD, (void*)y2F, TT, DD);
        residual_ln<<<BB * TT, 128, 0, stream>>>(encF, y2F, ng + i * DD, nb + i * DD,
                                                 encF, encB, TT);
    }

    // ---- length regulator ----
    cumsum_k<<<1, 32, 0, stream>>>(durs, cs);
    zero_guards<<<(BB * 2 * DD + 255) / 256, 256, 0, stream>>>(melB, MAXMEL, DD, BB * 2 * DD);
    zero_guards<<<(BB * 2 * C1 + 255) / 256, 256, 0, stream>>>(y1B, MAXMEL, C1, BB * 2 * C1);
    regulate<<<BB * MAXMEL, 128, 0, stream>>>(encF, cs, melF, melB);

    // ---- decoder: same 4 layers on mel ----
    for (int i = 0; i < LL; ++i) {
        dim3 g1(BB * MAXMEL / 128, C1 / 64);
        gemm_wmma<DD, C1, 3, 0><<<g1, 128, 0, stream>>>(
            melB, w1t + (size_t)i * 3 * C1 * DD, bc1 + i * C1, (void*)y1B, MAXMEL, C1);
        dim3 g2(BB * MAXMEL / 128, DD / 64);
        gemm_wmma<C1, DD, 3, 1><<<g2, 128, 0, stream>>>(
            y1B, w2t + (size_t)i * 3 * DD * C1, bc2 + i * DD, (void*)y2F, MAXMEL, DD);
        residual_ln<<<BB * MAXMEL, 128, 0, stream>>>(melF, y2F, ng + i * DD, nb + i * DD,
                                                     melF, melB, MAXMEL);
    }

    // ---- final projection mel @ Wout + bout -> d_out [B,1280,80] ----
    dim3 gf(BB * MAXMEL / 128, 2);   // N padded to 128, store guarded n<80
    gemm_wmma<DD, 128, 1, 2><<<gf, 128, 0, stream>>>(melB, wot, bout, (void*)outF, MAXMEL, NOUT);

    // ---- append durations (as float) ----
    copy_dur<<<(BB * TT + 255) / 256, 256, 0, stream>>>(durs, outF + (size_t)BB * MAXMEL * NOUT, BB * TT);
}
